// NonLocalAttention_73194832658568
// MI455X (gfx1250) — matrix operand order repaired and checked by reading, compiled, not run on hardware
//
#include <hip/hip_runtime.h>
#include <hip/hip_bf16.h>

// ---------------------------------------------------------------------------
// NonLocalAttention for MI455X (gfx1250), wave32 + WMMA 16x16x32 f16 path.
// v3: K blocks double-buffered in LDS via GLOBAL_LOAD_ASYNC_TO_LDS_B128
//     (ASYNCcnt pipeline), score B-fragments from LDS, rest as v2.
// ---------------------------------------------------------------------------

typedef _Float16 v16h __attribute__((ext_vector_type(16)));
typedef _Float16 v8h  __attribute__((ext_vector_type(8)));
typedef float    v8f  __attribute__((ext_vector_type(8)));

#define BN_EPS 1e-5f
#define CDIM   256
#define NSEQ   4096
#define NBATCH 4
#define MROWS  (NBATCH * NSEQ)   // 16384

static __device__ __forceinline__ v8f wmma_f16(v16h a, v16h b, v8f c) {
  // D = A(16x32 f16) * B(32x16 f16) + C(16x16 f32)
  return __builtin_amdgcn_wmma_f32_16x16x32_f16(
      /*neg_a=*/false, a, /*neg_b=*/false, b,
      /*c_mod=*/(short)0, c, /*reuse_a=*/false, /*reuse_b=*/false);
}

// ---- CDNA5 async copy: global -> LDS, 16 bytes per lane, ASYNCcnt-tracked.
static __device__ __forceinline__ void async_load_b128(void* lds, const void* g) {
  asm volatile("global_load_async_to_lds_b128 %0, %1, off"
               :
               : "v"((__attribute__((address_space(3))) char*)lds),
                 "v"((const __attribute__((address_space(1))) char*)g)
               : "memory");
}
static __device__ __forceinline__ void wait_async_le8() {
  asm volatile("s_wait_asynccnt 0x8" ::: "memory");
}
static __device__ __forceinline__ void wait_async_le0() {
  asm volatile("s_wait_asynccnt 0x0" ::: "memory");
}

// A fragment (16x32, f16) from a row-major matrix (global), origin (row0,k0).
// Layout: lanes 0-15 -> M=lane; elems 0..15 = K k0+hi*16.. (contiguous).
static __device__ __forceinline__ v16h load_a_g(const _Float16* __restrict__ A,
                                                int ld, int row0, int k0, int lane) {
  const _Float16* p = A + (size_t)(row0 + (lane & 15)) * ld + k0 + ((lane >> 4) << 4);
  v8h lo = *(const v8h*)(p);
  v8h hi = *(const v8h*)(p + 8);
  v16h a;
#pragma unroll
  for (int e = 0; e < 8; ++e) { a[e] = lo[e]; a[8 + e] = hi[e]; }
  return a;
}

// Same A layout from an LDS row-major tile with row stride ld (halfs).
static __device__ __forceinline__ v16h load_a_lds(const _Float16* P, int ld,
                                                  int k0, int lane) {
  const _Float16* p = P + (lane & 15) * ld + k0 + ((lane >> 4) << 4);
  v8h lo = *(const v8h*)(p);
  v8h hi = *(const v8h*)(p + 8);
  v16h a;
#pragma unroll
  for (int e = 0; e < 8; ++e) { a[e] = lo[e]; a[8 + e] = hi[e]; }
  return a;
}

// B fragment (32x16, f16) where B = S^T: B[k][n] = S[row0+n][d0+k].
// Per lane: 16 contiguous halfs -> one 32-byte load (2x global_load_b128).
static __device__ __forceinline__ v16h load_bT_g(const _Float16* __restrict__ S,
                                                 int ld, int row0, int d0, int lane) {
  const _Float16* p = S + (size_t)(row0 + (lane & 15)) * ld + d0 + ((lane >> 4) << 4);
  return *(const v16h*)p;
}

// Same B^T fragment pattern from an LDS row-major tile (2x ds_load_b128).
static __device__ __forceinline__ v16h load_bT_lds(const _Float16* S, int ld,
                                                   int row0, int d0, int lane) {
  const _Float16* p = S + (row0 + (lane & 15)) * ld + d0 + ((lane >> 4) << 4);
  v8h lo = *(const v8h*)(p);
  v8h hi = *(const v8h*)(p + 8);
  v16h b;
#pragma unroll
  for (int e = 0; e < 8; ++e) { b[e] = lo[e]; b[8 + e] = hi[e]; }
  return b;
}

// D fragment (16x16 f32) store: VGPR r -> row r (+8 for upper lane half).
template <typename T>
static __device__ __forceinline__ void store_d(T* __restrict__ O, int ld, int row0,
                                               int col0, int lane, v8f c) {
  int col = col0 + (lane & 15);
  int rb  = (lane >> 4) << 3;
#pragma unroll
  for (int r = 0; r < 8; ++r) O[(size_t)(row0 + rb + r) * ld + col] = (T)c[r];
}

// Transposed D store: O[col][row] with leading dim ldT; 8 contiguous elems/lane.
template <typename T>
static __device__ __forceinline__ void store_d_T(T* __restrict__ O, int ldT, int row0,
                                                 int col0, int lane, v8f c) {
  int col = col0 + (lane & 15);
  int rb  = (lane >> 4) << 3;
  T* p = O + (size_t)col * ldT + row0 + rb;
#pragma unroll
  for (int r = 0; r < 8; ++r) p[r] = (T)c[r];
}

// ---------------------------------------------------------------------------
// Kernel 1: depthwise 7x7 (SAME) + ReLU + BN(inference) -> f16 activations.
// ---------------------------------------------------------------------------
__global__ __launch_bounds__(256) void dwconv_bn_kernel(
    const float* __restrict__ x, const float* __restrict__ kern,  // (7,7,C,1)
    const float* __restrict__ gamma, const float* __restrict__ beta,
    const float* __restrict__ mean, const float* __restrict__ var,
    _Float16* __restrict__ y) {
  int idx = blockIdx.x * 256 + threadIdx.x;     // [0, 4*64*64*256)
  int c = idx & 255;
  int w = (idx >> 8) & 63;
  int h = (idx >> 14) & 63;
  int b = idx >> 20;
  float acc = 0.0f;
#pragma unroll
  for (int kh = 0; kh < 7; ++kh) {
    int hh = h + kh - 3;
    if (hh < 0 || hh > 63) continue;
#pragma unroll
    for (int kw = 0; kw < 7; ++kw) {
      int ww = w + kw - 3;
      if (ww < 0 || ww > 63) continue;
      acc += x[(((b * 64 + hh) * 64) + ww) * 256 + c] *
             kern[(kh * 7 + kw) * 256 + c];
    }
  }
  acc = fmaxf(acc, 0.0f);                                   // relu
  acc = (acc - mean[c]) * (gamma[c] * rsqrtf(var[c] + BN_EPS)) + beta[c];
  y[idx] = (_Float16)acc;
}

// ---------------------------------------------------------------------------
// Kernel 2: f32 -> f16 + transpose (256x256): d[n][k] = s[k][n].
// ---------------------------------------------------------------------------
__global__ __launch_bounds__(256) void cvt_f16_t_kernel(const float* __restrict__ s,
                                                        _Float16* __restrict__ d) {
  int i = blockIdx.x * 256 + threadIdx.x;       // [0, 65536)
  int k = i >> 8, n = i & 255;
  d[n * CDIM + k] = (_Float16)s[k * CDIM + n];
}

// ---------------------------------------------------------------------------
// Kernel 3/5: WMMA GEMM: out[M x 256] = op(A[M x 256] * WT^T)
// WT is the 256x256 weight stored transposed (WT[n][k] = W[k][n]).
// block = 128 threads (4 waves); wave w -> rows blockIdx.x*64 + 16w, 64 cols.
// ---------------------------------------------------------------------------
template <bool RELU, bool TRANSOUT, typename OutT>
__global__ __launch_bounds__(128) void gemm_wmma_kernel(
    const _Float16* __restrict__ A, const _Float16* __restrict__ WT,
    OutT* __restrict__ out, int M) {
  int wave = threadIdx.x >> 5;
  int lane = threadIdx.x & 31;
  int row0 = blockIdx.x * 64 + wave * 16;
  int col0 = blockIdx.y * 64;

  v8f acc[4] = {};
#pragma unroll
  for (int k0 = 0; k0 < CDIM; k0 += 32) {
    v16h a = load_a_g(A, CDIM, row0, k0, lane);
#pragma unroll
    for (int t = 0; t < 4; ++t) {
      v16h b = load_bT_g(WT, CDIM, col0 + 16 * t, k0, lane);
      acc[t] = wmma_f16(a, b, acc[t]);
    }
  }
#pragma unroll
  for (int t = 0; t < 4; ++t) {
    v8f o = acc[t];
    if (RELU) {
#pragma unroll
      for (int r = 0; r < 8; ++r) o[r] = fmaxf(o[r], 0.0f);
    }
    if (TRANSOUT) store_d_T<OutT>(out, M, row0, col0 + 16 * t, lane, o);
    else          store_d<OutT>(out, CDIM, row0, col0 + 16 * t, lane, o);
  }
}

// ---------------------------------------------------------------------------
// Kernel 4: flash attention. Block = 128 threads = 4 waves, one 16-query tile.
// Q tile staged once in LDS; K blocks (32 keys x 256 dims, 16 KB) are
// double-buffered in LDS via async global->LDS copies (ASYNCcnt pipeline).
// Each wave recomputes scores and owns a 64-dim slice of P*V (V^T layout).
// ---------------------------------------------------------------------------
__global__ __launch_bounds__(128) void flash_attn_kernel(
    const _Float16* __restrict__ Q, const _Float16* __restrict__ K,
    const _Float16* __restrict__ VT, const float* __restrict__ scale_p,
    _Float16* __restrict__ O) {
  __shared__ __align__(32) _Float16 Qs[16 * CDIM];      //  8 KB
  __shared__ __align__(32) _Float16 Ks[2][32 * CDIM];   // 32 KB (double buffer)
  __shared__ __align__(32) _Float16 Ps[4][16 * 32];     //  4 KB per-wave slabs

  int tid  = threadIdx.x;
  int wave = tid >> 5;
  int lane = tid & 31;
  int batch = blockIdx.x >> 8;          // 256 query tiles per batch
  int qt    = blockIdx.x & 255;
  int rowQ  = batch * NSEQ + qt * 16;
  int kvB   = batch * NSEQ;
  int ds    = wave * 64;                // this wave's output-dim slice
  float scale = scale_p[0];

  // Cooperative Q tile load: 4096 halfs, 8 halfs x 4 chunks per thread.
  {
    const _Float16* src = Q + (size_t)rowQ * CDIM;
#pragma unroll
    for (int i = 0; i < 4; ++i) {
      int off = (tid + i * 128) * 8;
      *(v8h*)(Qs + off) = *(const v8h*)(src + off);
    }
  }

  // Prologue: async-fill K buffer 0 (1024 x 16B chunks over 128 threads).
  {
    const char* src = (const char*)(K + (size_t)kvB * CDIM);
#pragma unroll
    for (int i = 0; i < 8; ++i) {
      int c = (tid + i * 128) * 16;
      async_load_b128((char*)Ks[0] + c, src + c);
    }
  }

  v8f acc[4] = {};
  float m[8], l[8];
#pragma unroll
  for (int r = 0; r < 8; ++r) { m[r] = -1e30f; l[r] = 0.0f; }

  _Float16* myPs = Ps[wave];
  int col = lane & 15;
  int rb  = (lane >> 4) << 3;

  for (int kb = 0; kb < NSEQ; kb += 32) {
    int cur = (kb >> 5) & 1;
    if (kb + 32 < NSEQ) {
      // Async prefetch next K block into the other buffer, then wait for the
      // current buffer (leave the 8 just-issued copies outstanding).
      const char* src = (const char*)(K + (size_t)(kvB + kb + 32) * CDIM);
      char* dst = (char*)Ks[cur ^ 1];
#pragma unroll
      for (int i = 0; i < 8; ++i) {
        int c = (tid + i * 128) * 16;
        async_load_b128(dst + c, src + c);
      }
      __builtin_prefetch(VT + (size_t)ds * MROWS + kvB + kb + 32, 0, 1);
      wait_async_le8();
    } else {
      wait_async_le0();
    }
    __syncthreads();   // current K buffer visible to all waves

    // S = Q * K^T for 32 keys -> two 16x16 f32 tiles (B-fragments from LDS).
    v8f s0 = {}, s1 = {};
#pragma unroll
    for (int d = 0; d < 8; ++d) {
      v16h a  = load_a_lds(Qs, CDIM, d * 32, lane);
      v16h b0 = load_bT_lds(Ks[cur], CDIM, 0,  d * 32, lane);
      v16h b1 = load_bT_lds(Ks[cur], CDIM, 16, d * 32, lane);
      s0 = wmma_f16(a, b0, s0);
      s1 = wmma_f16(a, b1, s1);
    }
    // Online softmax update; rows live in one VGPR across each 16-lane half.
#pragma unroll
    for (int r = 0; r < 8; ++r) {
      float v0 = (float)s0[r] * scale;
      float v1 = (float)s1[r] * scale;
      float mx = fmaxf(v0, v1);
      mx = fmaxf(mx, __shfl_xor(mx, 1, 32));
      mx = fmaxf(mx, __shfl_xor(mx, 2, 32));
      mx = fmaxf(mx, __shfl_xor(mx, 4, 32));
      mx = fmaxf(mx, __shfl_xor(mx, 8, 32));
      float mn   = fmaxf(m[r], mx);
      float corr = __expf(m[r] - mn);
      float e0 = __expf(v0 - mn);
      float e1 = __expf(v1 - mn);
      float rs = e0 + e1;
      rs += __shfl_xor(rs, 1, 32);
      rs += __shfl_xor(rs, 2, 32);
      rs += __shfl_xor(rs, 4, 32);
      rs += __shfl_xor(rs, 8, 32);
      l[r] = l[r] * corr + rs;
      m[r] = mn;
      // Write P straight into the per-wave LDS slab (D layout -> row major).
      myPs[(rb + r) * 32 + col]      = (_Float16)e0;
      myPs[(rb + r) * 32 + 16 + col] = (_Float16)e1;
#pragma unroll
      for (int t = 0; t < 4; ++t) acc[t][r] *= corr;
    }
    // Re-shape P through LDS (same-wave LDS ops are in-order) and do O += P*V.
    v16h pa = load_a_lds(myPs, 32, 0, lane);
#pragma unroll
    for (int t = 0; t < 4; ++t) {
      v16h bv = load_bT_g(VT, MROWS, ds + 16 * t, kvB + kb, lane);
      acc[t] = wmma_f16(pa, bv, acc[t]);
    }
    __syncthreads();   // consumption done before next overwrite of Ks[cur^1]
  }
  // Normalize and emit f16 attention output (row-major [16384][256]).
#pragma unroll
  for (int t = 0; t < 4; ++t) {
    v8f o = acc[t];
#pragma unroll
    for (int r = 0; r < 8; ++r) o[r] = o[r] / l[r];
    store_d<_Float16>(O, CDIM, rowQ, ds + 16 * t, lane, o);
  }
}

// ---------------------------------------------------------------------------
// Launch: dwconv+bn -> cvt/transpose weights -> q/k/v GEMMs -> attn -> out.
// ---------------------------------------------------------------------------
extern "C" void kernel_launch(void* const* d_in, const int* in_sizes, int n_in,
                              void* d_out, int out_size, void* d_ws, size_t ws_size,
                              hipStream_t stream) {
  (void)in_sizes; (void)n_in; (void)out_size; (void)ws_size;
  const float* x     = (const float*)d_in[0];
  const float* dwk   = (const float*)d_in[1];
  const float* gamma = (const float*)d_in[2];
  const float* beta  = (const float*)d_in[3];
  const float* mean  = (const float*)d_in[4];
  const float* var   = (const float*)d_in[5];
  const float* wq    = (const float*)d_in[6];
  const float* wk    = (const float*)d_in[7];
  const float* wv    = (const float*)d_in[8];
  const float* wo    = (const float*)d_in[9];
  const float* scale = (const float*)d_in[10];
  float* out = (float*)d_out;

  char* ws = (char*)d_ws;
  size_t o = 0;
  _Float16* y16  = (_Float16*)(ws + o); o += (size_t)MROWS * CDIM * 2;  // 8 MB
  _Float16* wqT  = (_Float16*)(ws + o); o += CDIM * CDIM * 2;
  _Float16* wkT  = (_Float16*)(ws + o); o += CDIM * CDIM * 2;
  _Float16* wvT  = (_Float16*)(ws + o); o += CDIM * CDIM * 2;
  _Float16* woT  = (_Float16*)(ws + o); o += CDIM * CDIM * 2;
  _Float16* q16  = (_Float16*)(ws + o); o += (size_t)MROWS * CDIM * 2;
  _Float16* k16  = (_Float16*)(ws + o); o += (size_t)MROWS * CDIM * 2;
  _Float16* vT16 = (_Float16*)(ws + o); o += (size_t)MROWS * CDIM * 2;  // V^T
  _Float16* a16  = (_Float16*)(ws + o); o += (size_t)MROWS * CDIM * 2;  // ~41 MB

  dwconv_bn_kernel<<<MROWS, 256, 0, stream>>>(x, dwk, gamma, beta, mean, var, y16);

  cvt_f16_t_kernel<<<256, 256, 0, stream>>>(wq, wqT);
  cvt_f16_t_kernel<<<256, 256, 0, stream>>>(wk, wkT);
  cvt_f16_t_kernel<<<256, 256, 0, stream>>>(wv, wvT);
  cvt_f16_t_kernel<<<256, 256, 0, stream>>>(wo, woT);

  dim3 gg(MROWS / 64, CDIM / 64);                    // (256, 4)
  gemm_wmma_kernel<false, false, _Float16><<<gg, 128, 0, stream>>>(y16, wqT, q16, MROWS);
  gemm_wmma_kernel<true,  false, _Float16><<<gg, 128, 0, stream>>>(y16, wkT, k16, MROWS);
  gemm_wmma_kernel<false, true,  _Float16><<<gg, 128, 0, stream>>>(y16, wvT, vT16, MROWS);

  flash_attn_kernel<<<NBATCH * (NSEQ / 16), 128, 0, stream>>>(q16, k16, vT16,
                                                              scale, a16);

  gemm_wmma_kernel<false, false, float><<<gg, 128, 0, stream>>>(a16, woT, out, MROWS);
}
